// HyperSETAlterAttention_71614284694168
// MI455X (gfx1250) — matrix operand order repaired and checked by reading, not compiled, not run on hardware
//
#include <hip/hip_runtime.h>
#include <hip/hip_bf16.h>
#include <math.h>

typedef __bf16 v16bf __attribute__((ext_vector_type(16)));
typedef __bf16 v8bf  __attribute__((ext_vector_type(8)));
typedef float  v8f   __attribute__((ext_vector_type(8)));
typedef unsigned int v4u __attribute__((ext_vector_type(4)));
typedef int v8i __attribute__((ext_vector_type(8)));
typedef int v4i __attribute__((ext_vector_type(4)));

#define HEADS 12
#define DH    64
#define TT    2048
#define CC    768
#define BBATCH 2
#define MROWS (BBATCH*TT)      // 4096
#define EPSV  1.1920929e-7f    // FLT_EPSILON

// ---------------------------------------------------------------------------
// WMMA fragment helpers (layouts per CDNA5 ISA 7.12.2, wave32)
// ---------------------------------------------------------------------------
__device__ __forceinline__ v16bf frag_from(const __bf16* p0, const __bf16* p1) {
    v8bf lo = *(const v8bf*)p0;
    v8bf hi = *(const v8bf*)p1;
    v16bf r;
#pragma unroll
    for (int i = 0; i < 8; ++i) { r[i] = lo[i]; r[8 + i] = hi[i]; }
    return r;
}

// A fragment (16 x 32, bf16) from row-major [rows, ldk]
__device__ __forceinline__ v16bf load_A(const __bf16* base, int ldk, int k0) {
    int lane = threadIdx.x & 31;
    int h = lane >> 4, m = lane & 15;
    const __bf16* p = base + (size_t)m * ldk + k0 + h * 8;
    return frag_from(p, p + 16);
}

// B fragment (32 x 16, bf16) from B^T stored row-major [16 rows = N, ldk]
__device__ __forceinline__ v16bf load_B(const __bf16* baseT, int ldk, int k0) {
    int lane = threadIdx.x & 31;
    int h = lane >> 4, n = lane & 15;
    const __bf16* p = baseT + (size_t)n * ldk + k0 + h * 8;
    return frag_from(p, p + 16);
}

__device__ __forceinline__ v8f wmma_bf16(v16bf a, v16bf b, v8f c) {
    return __builtin_amdgcn_wmma_f32_16x16x32_bf16(false, a, false, b, (short)0, c,
                                                   false, false);
}

// ---------------------------------------------------------------------------
// Tensor Data Mover: stage a [16 rows x width bf16] tile (row stride in elems)
// from global into LDS.  D# layout per CDNA5 ISA.
// 6-arg builtin form (clang-23 / therock-10.0).
// ---------------------------------------------------------------------------
__device__ __forceinline__ void tdm_load_tile16(const __bf16* gsrc,
                                                const __bf16* lds_dst,
                                                int width, int row_stride) {
    unsigned long long ga = (unsigned long long)(size_t)gsrc;
    unsigned lds_byte = (unsigned)(size_t)lds_dst;   // flat LDS addr truncates to offset
    v4u g0;
    g0.x = 1u;                                            // count=1, user mode
    g0.y = lds_byte;                                      // lds_addr
    g0.z = (unsigned)ga;                                  // global_addr[31:0]
    g0.w = (unsigned)((ga >> 32) & 0x1FFFFFFull) | (2u << 30);  // addr[56:32] | type=2
    v8i g1;
    g1[0] = (1 << 16);                                    // data_size = 2 bytes
    g1[1] = (width & 0xffff) << 16;                       // tensor_dim0[15:0]
    g1[2] = (16 << 16);                                   // tensor_dim1[15:0] = 16 rows
    g1[3] = (width & 0xffff) << 16;                       // tile_dim0 = width
    g1[4] = 16;                                           // tile_dim1 = 16, tile_dim2 = 0
    g1[5] = row_stride;                                   // tensor_dim0_stride[31:0]
    g1[6] = 0;
    g1[7] = 0;
    v4i z4 = {0, 0, 0, 0};
    v8i z8 = {0, 0, 0, 0, 0, 0, 0, 0};
    __builtin_amdgcn_tensor_load_to_lds(g0, g1, z4, z4, z8, 0);
}

// ---------------------------------------------------------------------------
// 1) prep: bf16 conversions of x, W, and W^T
// ---------------------------------------------------------------------------
__global__ void k_prep(const float* __restrict__ x, const float* __restrict__ W,
                       __bf16* __restrict__ x_bf, __bf16* __restrict__ W_bf,
                       __bf16* __restrict__ Wt_bf) {
    int idx = blockIdx.x * 256 + threadIdx.x;
    if (idx < MROWS * CC) x_bf[idx] = (__bf16)x[idx];
    if (idx < CC * CC) {
        W_bf[idx] = (__bf16)W[idx];
        int o = idx / CC, c = idx % CC;
        Wt_bf[idx] = (__bf16)W[c * CC + o];
    }
}

// ---------------------------------------------------------------------------
// 2) WMMA GEMM: out(f32)[M,N] = A[M,K] * (Bt row-major [N,K])^T
//    one wave per 16x64 output strip: A fragment reused across 4 WMMAs
// ---------------------------------------------------------------------------
__global__ void k_gemm_f32out(const __bf16* __restrict__ A,
                              const __bf16* __restrict__ Bt,
                              float* __restrict__ out, int M, int N, int K) {
    int wid = (blockIdx.x * blockDim.x + threadIdx.x) >> 5;
    int nstrips = N >> 6;
    int mt = wid / nstrips, nt4 = wid % nstrips;
    const __bf16* Arow  = A  + (size_t)mt * 16 * K;
    const __bf16* Btrow = Bt + (size_t)nt4 * 64 * K;
    v8f acc[4] = {};
    for (int k0 = 0; k0 < K; k0 += 32) {
        if (k0 + 32 < K)
            __builtin_prefetch(Arow + (size_t)(threadIdx.x & 15) * K + k0 + 32, 0, 0);
        v16bf a = load_A(Arow, K, k0);
#pragma unroll
        for (int n = 0; n < 4; ++n) {
            v16bf b = load_B(Btrow + (size_t)n * 16 * K, K, k0);
            acc[n] = wmma_bf16(a, b, acc[n]);
        }
    }
    int lane = threadIdx.x & 31, h = lane >> 4, nl = lane & 15;
#pragma unroll
    for (int n = 0; n < 4; ++n)
#pragma unroll
        for (int r = 0; r < 8; ++r)
            out[(size_t)(mt * 16 + h * 8 + r) * N + nt4 * 64 + n * 16 + nl] = acc[n][r];
}

// ---------------------------------------------------------------------------
// 3) qkv: per (b,t,h) row of dh=64, RMSNorm; q gets 1/sqrt(dh) folded in.
//    q,k stored [BH, T, DH]; v stored transposed [BH, DH, T].
// ---------------------------------------------------------------------------
__global__ void k_qkv(const float* __restrict__ w, const float* __restrict__ g1,
                      const float* __restrict__ g2, const float* __restrict__ g3,
                      __bf16* __restrict__ q, __bf16* __restrict__ k,
                      __bf16* __restrict__ vT) {
    int wid  = (blockIdx.x * blockDim.x + threadIdx.x) >> 5;
    int lane = threadIdx.x & 31;
    int bt = wid / HEADS, h = wid % HEADS;
    int b = bt / TT, t = bt % TT;
    const float* wrow = w + (size_t)bt * CC + h * DH;
    float2 val = *(const float2*)(wrow + lane * 2);
    float ss = val.x * val.x + val.y * val.y;
#pragma unroll
    for (int m = 1; m < 32; m <<= 1) ss += __shfl_xor(ss, m, 32);
    float sc = rsqrtf(ss * (1.0f / DH) + EPSV);
    int bh = b * HEADS + h;
    int c0 = lane * 2;
    size_t qo = (size_t)bh * TT * DH + (size_t)t * DH + c0;
    q[qo]     = (__bf16)(val.x * sc * g1[c0]     * 0.125f);
    q[qo + 1] = (__bf16)(val.y * sc * g1[c0 + 1] * 0.125f);
    k[qo]     = (__bf16)(val.x * sc * g2[c0]);
    k[qo + 1] = (__bf16)(val.y * sc * g2[c0 + 1]);
    size_t vo = (size_t)bh * DH * TT + (size_t)c0 * TT + t;
    vT[vo]      = (__bf16)(val.x * sc * g3[c0]);
    vT[vo + TT] = (__bf16)(val.y * sc * g3[c0 + 1]);
}

// ---------------------------------------------------------------------------
// 4) stats: online softmax row max/sum over S = A * Bt^T tiles.
//    B tiles staged to LDS by the Tensor Data Mover, double buffered.
// ---------------------------------------------------------------------------
__global__ void k_stats(const __bf16* __restrict__ A, const __bf16* __restrict__ Bt,
                        float* __restrict__ om, float* __restrict__ os) {
    __shared__ __align__(16) __bf16 kbuf[4][2][16 * DH];
    int bh = blockIdx.y;
    int wv_in = threadIdx.x >> 5;
    int m0 = (blockIdx.x * 4 + wv_in) * 16;
    int lane = threadIdx.x & 31, h = lane >> 4, nl = lane & 15;
    const __bf16* Ab = A  + (size_t)bh * TT * DH + (size_t)m0 * DH;
    const __bf16* Bb = Bt + (size_t)bh * TT * DH;
    v16bf a0 = load_A(Ab, DH, 0);
    v16bf a1 = load_A(Ab, DH, 32);
    float mx[8], sm[8];
#pragma unroll
    for (int r = 0; r < 8; ++r) { mx[r] = -INFINITY; sm[r] = 0.0f; }

    const int NT = TT / 16;
    tdm_load_tile16(Bb, kbuf[wv_in][0], DH, DH);
    for (int ct = 0; ct < NT; ++ct) {
        if (ct + 1 < NT) {
            tdm_load_tile16(Bb + (size_t)(ct + 1) * 16 * DH, kbuf[wv_in][(ct + 1) & 1],
                            DH, DH);
            __builtin_amdgcn_s_wait_tensorcnt(1);
        } else {
            __builtin_amdgcn_s_wait_tensorcnt(0);
        }
        const __bf16* Kt = kbuf[wv_in][ct & 1];
        v8f s = {};
        s = wmma_bf16(a0, load_B(Kt, DH, 0), s);
        s = wmma_bf16(a1, load_B(Kt, DH, 32), s);
#pragma unroll
        for (int r = 0; r < 8; ++r) {
            float v = s[r];
            float tmx = v;
#pragma unroll
            for (int m = 1; m < 16; m <<= 1) tmx = fmaxf(tmx, __shfl_xor(tmx, m, 32));
            float nm = fmaxf(mx[r], tmx);
            float e = __expf(v - nm);
#pragma unroll
            for (int m = 1; m < 16; m <<= 1) e += __shfl_xor(e, m, 32);
            sm[r] = sm[r] * __expf(mx[r] - nm) + e;
            mx[r] = nm;
        }
    }
#pragma unroll
    for (int r = 0; r < 8; ++r) {
        if (nl == r) {
            om[(size_t)bh * TT + m0 + h * 8 + r] = mx[r];
            os[(size_t)bh * TT + m0 + h * 8 + r] = sm[r];
        }
    }
}

// ---------------------------------------------------------------------------
// 5) attn*v: recompute S tiles (K tiles via TDM), a = exp(S-rm)/rs + exp(S-cm)/cs,
//    y += a @ v with `a` re-fragmented through per-wave LDS.
// ---------------------------------------------------------------------------
__global__ void k_attnv(const __bf16* __restrict__ q, const __bf16* __restrict__ k,
                        const __bf16* __restrict__ vT,
                        const float* __restrict__ rm, const float* __restrict__ rs,
                        const float* __restrict__ cm, const float* __restrict__ cs,
                        __bf16* __restrict__ y) {
    __shared__ __align__(16) __bf16 abuf[4][16][32];
    __shared__ __align__(16) __bf16 kbuf[4][2][16 * DH];
    int bh = blockIdx.y;
    int b = bh / HEADS, hh = bh % HEADS;
    int wv_in = threadIdx.x >> 5;
    int m0 = (blockIdx.x * 4 + wv_in) * 16;
    int lane = threadIdx.x & 31, h = lane >> 4, nl = lane & 15;

    const __bf16* Ab = q  + (size_t)bh * TT * DH + (size_t)m0 * DH;
    const __bf16* Kb = k  + (size_t)bh * TT * DH;
    const __bf16* Vb = vT + (size_t)bh * DH * TT;
    v16bf qa0 = load_A(Ab, DH, 0);
    v16bf qa1 = load_A(Ab, DH, 32);

    float rmv[8], irs[8];
#pragma unroll
    for (int r = 0; r < 8; ++r) {
        rmv[r] = rm[(size_t)bh * TT + m0 + h * 8 + r];
        irs[r] = 1.0f / rs[(size_t)bh * TT + m0 + h * 8 + r];
    }

    v8f accy[4] = {};
    const int NT = TT / 16;
    tdm_load_tile16(Kb, kbuf[wv_in][0], DH, DH);
    for (int ct = 0; ct < NT; ++ct) {
        if (ct + 1 < NT) {
            tdm_load_tile16(Kb + (size_t)(ct + 1) * 16 * DH, kbuf[wv_in][(ct + 1) & 1],
                            DH, DH);
            __builtin_amdgcn_s_wait_tensorcnt(1);
        } else {
            __builtin_amdgcn_s_wait_tensorcnt(0);
        }
        const __bf16* Kt = kbuf[wv_in][ct & 1];
        v8f s = {};
        s = wmma_bf16(qa0, load_B(Kt, DH, 0), s);
        s = wmma_bf16(qa1, load_B(Kt, DH, 32), s);
        int col = ct * 16 + nl;
        float cmv  = cm[(size_t)bh * TT + col];
        float icsv = 1.0f / cs[(size_t)bh * TT + col];
#pragma unroll
        for (int r = 0; r < 8; ++r) {
            float av = __expf(s[r] - rmv[r]) * irs[r] + __expf(s[r] - cmv) * icsv;
            abuf[wv_in][h * 8 + r][(ct & 1) * 16 + nl] = (__bf16)av;
        }
        if (ct & 1) {   // 32 columns of `a` staged: K=32 chunk of a@v
            const __bf16* pm = &abuf[wv_in][0][0] + (size_t)(lane & 15) * 32;
            v16bf af = frag_from(pm + h * 8, pm + 16 + h * 8);
            int j0 = (ct >> 1) * 32;
#pragma unroll
            for (int f = 0; f < 4; ++f) {
                const __bf16* bp = Vb + (size_t)(f * 16) * TT + j0;
                accy[f] = wmma_bf16(af, load_B(bp, TT, 0), accy[f]);
            }
        }
    }
#pragma unroll
    for (int f = 0; f < 4; ++f)
#pragma unroll
        for (int r = 0; r < 8; ++r)
            y[((size_t)b * TT + m0 + h * 8 + r) * CC + hh * DH + f * 16 + nl] =
                (__bf16)accy[f][r];
}

// ---------------------------------------------------------------------------
// launch
// ---------------------------------------------------------------------------
extern "C" void kernel_launch(void* const* d_in, const int* in_sizes, int n_in,
                              void* d_out, int out_size, void* d_ws, size_t ws_size,
                              hipStream_t stream) {
    const float* x  = (const float*)d_in[0];
    const float* W  = (const float*)d_in[1];
    const float* g1 = (const float*)d_in[2];
    const float* g2 = (const float*)d_in[3];
    const float* g3 = (const float*)d_in[4];
    float* out = (float*)d_out;

    char* p = (char*)d_ws;
    auto bump = [&](size_t bytes) {
        char* r = p;
        p += (bytes + 255) & ~(size_t)255;
        return r;
    };
    __bf16* x_bf  = (__bf16*)bump((size_t)MROWS * CC * 2);
    __bf16* W_bf  = (__bf16*)bump((size_t)CC * CC * 2);
    __bf16* Wt_bf = (__bf16*)bump((size_t)CC * CC * 2);
    float*  w_f32 = (float*) bump((size_t)MROWS * CC * 4);
    __bf16* q_bf  = (__bf16*)bump((size_t)BBATCH * HEADS * TT * DH * 2);
    __bf16* k_bf  = (__bf16*)bump((size_t)BBATCH * HEADS * TT * DH * 2);
    __bf16* vT_bf = (__bf16*)bump((size_t)BBATCH * HEADS * DH * TT * 2);
    __bf16* y_bf  = (__bf16*)bump((size_t)MROWS * CC * 2);
    float*  rm  = (float*)bump((size_t)BBATCH * HEADS * TT * 4);
    float*  rs  = (float*)bump((size_t)BBATCH * HEADS * TT * 4);
    float*  cmv = (float*)bump((size_t)BBATCH * HEADS * TT * 4);
    float*  csv = (float*)bump((size_t)BBATCH * HEADS * TT * 4);

    // 1) precision prep
    k_prep<<<(MROWS * CC + 255) / 256, 256, 0, stream>>>(x, W, x_bf, W_bf, Wt_bf);

    // 2) w = x @ W^T   (one wave per 16x64 strip, 8 waves/block)
    k_gemm_f32out<<<(MROWS / 16) * (CC / 64) / 8, 256, 0, stream>>>(
        x_bf, W_bf, w_f32, MROWS, CC, CC);

    // 3) rmsnorm -> q (pre-scaled by 1/sqrt(dh)), k, v^T
    k_qkv<<<(MROWS * HEADS) / 8, 256, 0, stream>>>(w_f32, g1, g2, g3, q_bf, k_bf, vT_bf);

    // 4) row stats then column stats (S^T pass; scale folded into q)
    dim3 gs(TT / 16 / 4, BBATCH * HEADS);
    k_stats<<<gs, 128, 0, stream>>>(q_bf, k_bf, rm, rs);
    k_stats<<<gs, 128, 0, stream>>>(k_bf, q_bf, cmv, csv);

    // 5) y = (softmax_r + softmax_c) @ v
    k_attnv<<<gs, 128, 0, stream>>>(q_bf, k_bf, vT_bf, rm, rs, cmv, csv, y_bf);

    // 6) out = y @ W
    k_gemm_f32out<<<(MROWS / 16) * (CC / 64) / 8, 256, 0, stream>>>(
        y_bf, Wt_bf, out, MROWS, CC, CC);
}
